// GraphRecommendationModel_48567490183265
// MI455X (gfx1250) — matrix operand order repaired and checked by reading, not compiled, not run on hardware
//
#include <hip/hip_runtime.h>
#include <hip/hip_bf16.h>

// ---------------------------------------------------------------------------
// GCN 2-layer forward for MI455X (gfx1250, wave32).
// GEMMs use v_wmma_f32_16x16x32_bf16 (bf16 in, f32 accumulate), register-
// blocked 4 row-tiles per block so B fragments are loaded once per wave.
// Aggregation = L2-resident float atomics (76.8MB node matrix < 192MB L2).
// ---------------------------------------------------------------------------

typedef __attribute__((ext_vector_type(16))) __bf16 v16bf;
typedef __attribute__((ext_vector_type(8)))  __bf16 v8bf;
typedef __attribute__((ext_vector_type(8)))  float  v8f;

#define N_USERS 50000
#define NN      150000        // total nodes (divisible by 16)
#define DD      128
#define EE      500000
#define BB      16384
#define BN_EPS  1e-5f
#define NTILES  9375          // NN/16 row tiles
#define TPB     4             // row tiles per block (register-blocked)

// ---- degree / symmetric normalization -------------------------------------
__global__ void k_init_deg(float* deg) {
  int i = blockIdx.x * blockDim.x + threadIdx.x;
  if (i < NN) deg[i] = 1.0f;                       // +1 self-loop
}
__global__ void k_count_deg(const int* __restrict__ edst, float* deg) {
  int e = blockIdx.x * blockDim.x + threadIdx.x;
  if (e < EE) atomicAdd(&deg[edst[e]], 1.0f);
}
__global__ void k_rsqrt_deg(float* deg) {
  int i = blockIdx.x * blockDim.x + threadIdx.x;
  if (i < NN) deg[i] = rsqrtf(deg[i]);             // deg^{-1/2}, in place
}

// ---- weight pack: fp32 row-major [K=128][N=128] -> WMMA B-fragment layout --
// Wp[((kt*8+nt)*32+lane)*16 + j] = bf16( W[(kt*32 + (j>=8)*16 + (lane>=16)*8 + (j&7))*128 + nt*16 + (lane&15)] )
__global__ void k_pack_w(const float* __restrict__ W, __bf16* __restrict__ Wp) {
  int o = blockIdx.x * blockDim.x + threadIdx.x;
  if (o >= DD * DD) return;
  int j    =  o        & 15;
  int lane = (o >> 4)  & 31;
  int nt   = (o >> 9)  & 7;
  int kt   = (o >> 12) & 3;
  int k = kt * 32 + ((j >> 3) << 4) + ((lane >> 4) << 3) + (j & 7);
  int n = nt * 16 + (lane & 15);
  Wp[o] = (__bf16)W[k * DD + n];
}

// ---- concat(user_emb, book_emb) -> bf16 row-major activations --------------
__global__ void k_concat_bf16(const float* __restrict__ ue, const float* __restrict__ be,
                              __bf16* __restrict__ hbf) {
  long long t = (long long)blockIdx.x * blockDim.x + threadIdx.x;
  if (t >= (long long)NN * DD) return;
  const long long ub = (long long)N_USERS * DD;
  float v = (t < ub) ? ue[t] : be[t - ub];
  hbf[t] = (__bf16)v;
}

// ---- GEMM: C[N,128] = A[N,128] (bf16) x Wp (bf16 packed), f32 out ----------
// block = 256 threads = 8 waves; wave nt owns one 16-column tile.
// Register-blocked: each block covers TPB=4 row tiles (64 rows); the wave's
// four B fragments (full K=128) are loaded once and reused -> 16 WMMA/wave.
__global__ void __launch_bounds__(256) k_gemm_wmma(const __bf16* __restrict__ A,
                                                   const __bf16* __restrict__ Wp,
                                                   float* __restrict__ C) {
  const int lane = threadIdx.x & 31;
  const int nt   = threadIdx.x >> 5;          // 0..7: output column tile
  const int m    = lane & 15;                 // A row within tile
  const int kq   = lane >> 4;                 // K-half select (ISA A layout)
  const int tileBase = blockIdx.x * TPB;

  // B fragments for this column tile, all 4 K-steps, held in registers.
  const __bf16* bptr = Wp + ((size_t)nt * 32 + lane) * 16;
  const v16bf bf0 = *(const v16bf*)(bptr + 0 * 4096);
  const v16bf bf1 = *(const v16bf*)(bptr + 1 * 4096);
  const v16bf bf2 = *(const v16bf*)(bptr + 2 * 4096);
  const v16bf bf3 = *(const v16bf*)(bptr + 3 * 4096);

#pragma unroll
  for (int t = 0; t < TPB; ++t) {
    const int rowTile = tileBase + t;
    if (rowTile >= NTILES) break;             // block-uniform -> scalar branch

    const __bf16* arow = A + (size_t)(rowTile * 16 + m) * DD + kq * 8;
    if (t + 1 < TPB && rowTile + 1 < NTILES)
      __builtin_prefetch((const void*)(arow + 16 * DD), 0, 3); // next tile's A

    v8f acc = {0.f, 0.f, 0.f, 0.f, 0.f, 0.f, 0.f, 0.f};
#pragma unroll
    for (int kt = 0; kt < 4; ++kt) {
      v8bf alo = *(const v8bf*)(arow + kt * 32);       // K = kt*32 + kq*8 + [0..7]
      v8bf ahi = *(const v8bf*)(arow + kt * 32 + 16);  // K = +16
      v16bf a = __builtin_shufflevector(alo, ahi,
          0, 1, 2, 3, 4, 5, 6, 7, 8, 9, 10, 11, 12, 13, 14, 15);
      const v16bf b = (kt == 0) ? bf0 : (kt == 1) ? bf1 : (kt == 2) ? bf2 : bf3;
      acc = __builtin_amdgcn_wmma_f32_16x16x32_bf16(
          /*neg_a=*/false, a, /*neg_b=*/false, b,
          /*c_mod=*/(short)0, acc, /*reuse_a=*/false, /*reuse_b=*/false);
    }

    // C/D layout: lane&15 -> N, VGPR r -> M = kq*8 + r
    const int col  = nt * 16 + (lane & 15);
    const int mrow = rowTile * 16 + kq * 8;
    float* cptr = C + (size_t)mrow * DD + col;
#pragma unroll
    for (int r = 0; r < 8; ++r) cptr[(size_t)r * DD] = acc[r];
  }
}

// ---- aggregation -----------------------------------------------------------
__global__ void k_agg_init(const float* __restrict__ hlin, const float* __restrict__ dis,
                           const float* __restrict__ bias, float* __restrict__ agg) {
  long long t = (long long)blockIdx.x * blockDim.x + threadIdx.x;
  if (t >= (long long)NN * DD) return;
  int n = (int)(t >> 7);
  int d = (int)(t & 127);
  float s = dis[n];
  agg[t] = hlin[t] * s * s + bias[d];              // self-loop (deg^-1) + bias
}

__global__ void k_agg_scatter(const int* __restrict__ esrc, const int* __restrict__ edst,
                              const float* __restrict__ hlin, const float* __restrict__ dis,
                              float* __restrict__ agg) {
  long long t = (long long)blockIdx.x * blockDim.x + threadIdx.x;
  if (t >= (long long)EE * 32) return;
  int e = (int)(t >> 5);
  int q = ((int)t & 31) << 2;                      // 32 lanes x 4 floats = 128 dims
  int s = esrc[e], d = edst[e];
  float norm = dis[s] * dis[d];
  const float4 v = *(const float4*)(hlin + (size_t)s * DD + q);
  float* ap = agg + (size_t)d * DD + q;
  atomicAdd(ap + 0, v.x * norm);
  atomicAdd(ap + 1, v.y * norm);
  atomicAdd(ap + 2, v.z * norm);
  atomicAdd(ap + 3, v.w * norm);
}

// ---- batch-norm ------------------------------------------------------------
__global__ void k_zero256(float* p) {
  if (threadIdx.x < 256) p[threadIdx.x] = 0.f;
}
__global__ void k_bn_stats(const float* __restrict__ x, float* __restrict__ gsum,
                           float* __restrict__ gsq) {
  int c = threadIdx.x;                             // 128 columns
  float s = 0.f, q = 0.f;
  for (int r = blockIdx.x; r < NN; r += gridDim.x) {
    float v = x[(size_t)r * DD + c];
    s += v; q += v * v;
  }
  atomicAdd(&gsum[c], s);
  atomicAdd(&gsq[c],  q);
}
__global__ void k_bn_final(const float* __restrict__ gsum, const float* __restrict__ gsq,
                           const float* __restrict__ g, const float* __restrict__ be,
                           float* __restrict__ scale, float* __restrict__ shift) {
  int c = threadIdx.x;
  if (c < DD) {
    float mean = gsum[c] * (1.0f / NN);
    float var  = gsq[c] * (1.0f / NN) - mean * mean;
    float sc   = g[c] * rsqrtf(var + BN_EPS);
    scale[c] = sc;
    shift[c] = be[c] - mean * sc;
  }
}
// fused BN+ReLU, emitting bf16 (next layer's WMMA A operand)
__global__ void k_bn_relu_bf16(const float* __restrict__ x, const float* __restrict__ scale,
                               const float* __restrict__ shift, __bf16* __restrict__ out) {
  long long t = (long long)blockIdx.x * blockDim.x + threadIdx.x;
  if (t >= (long long)NN * DD) return;
  int d = (int)(t & 127);
  float v = fmaxf(x[t] * scale[d] + shift[d], 0.f);
  out[t] = (__bf16)v;
}
// fused BN+ReLU, f32 in-place (final features for the rating dot-products)
__global__ void k_bn_relu_f32(const float* __restrict__ x, const float* __restrict__ scale,
                              const float* __restrict__ shift, float* __restrict__ out) {
  long long t = (long long)blockIdx.x * blockDim.x + threadIdx.x;
  if (t >= (long long)NN * DD) return;
  int d = (int)(t & 127);
  out[t] = fmaxf(x[t] * scale[d] + shift[d], 0.f);
}

// ---- ratings: one wave32 per pair ------------------------------------------
__global__ void k_ratings(const int* __restrict__ srcn, const int* __restrict__ dstn,
                          const float* __restrict__ h, float* __restrict__ out) {
  int t = blockIdx.x * blockDim.x + threadIdx.x;
  int w = t >> 5, lane = t & 31;
  if (w >= BB) return;
  int s = srcn[w], d = dstn[w];
  const float4 a = *(const float4*)(h + (size_t)s * DD + lane * 4);
  const float4 b = *(const float4*)(h + (size_t)d * DD + lane * 4);
  float p = a.x * b.x + a.y * b.y + a.z * b.z + a.w * b.w;
#pragma unroll
  for (int off = 16; off > 0; off >>= 1) p += __shfl_xor(p, off, 32);
  if (lane == 0) out[w] = p;
}

// ---------------------------------------------------------------------------
extern "C" void kernel_launch(void* const* d_in, const int* in_sizes, int n_in,
                              void* d_out, int out_size, void* d_ws, size_t ws_size,
                              hipStream_t stream) {
  (void)in_sizes; (void)n_in; (void)out_size; (void)ws_size;

  const int*   edge = (const int*)d_in[0];
  const int*   esrc = edge;
  const int*   edst = edge + EE;
  const int*   srcn = (const int*)d_in[2];
  const int*   dstn = (const int*)d_in[3];
  const float* uemb = (const float*)d_in[4];
  const float* bemb = (const float*)d_in[5];
  const float* W1   = (const float*)d_in[6];
  const float* b1   = (const float*)d_in[7];
  const float* g1   = (const float*)d_in[8];
  const float* be1  = (const float*)d_in[9];
  const float* W2   = (const float*)d_in[10];
  const float* b2   = (const float*)d_in[11];
  const float* g2   = (const float*)d_in[12];
  const float* be2  = (const float*)d_in[13];
  float* out = (float*)d_out;

  // workspace carve-up (all offsets 256B aligned); total ~192.7 MB
  char* ws = (char*)d_ws;
  size_t o = 0;
  __bf16* hbf  = (__bf16*)(ws + o); o += (size_t)NN * DD * 2;   // 38,400,000
  float*  hlin = (float*)(ws + o);  o += (size_t)NN * DD * 4;   // 76,800,000
  float*  agg  = (float*)(ws + o);  o += (size_t)NN * DD * 4;   // 76,800,000
  float*  dis  = (float*)(ws + o);  o += 600064;                // deg -> deg^-1/2
  __bf16* Wp   = (__bf16*)(ws + o); o += 2 * (size_t)DD * DD * 2; // packed W1|W2
  float*  stats = (float*)(ws + o); o += 2048;
  float* gsum  = stats;        // 128
  float* gsq   = stats + 128;  // 128
  float* scale = stats + 256;  // 128
  float* shift = stats + 384;  // 128

  const int T = 256;
  const long long ND = (long long)NN * DD;
  const int gND   = (int)((ND + T - 1) / T);        // 75000
  const int gN    = (NN + T - 1) / T;               // 586
  const int gE    = (EE + T - 1) / T;               // 1954
  const int gGemm = (NTILES + TPB - 1) / TPB;       // 2344 blocks, 64 rows each
  const int gScat = (int)(((long long)EE * 32) / T);// 62500 (exact)
  const int gRate = (BB * 32) / T;                  // 2048 (exact)

  // degree / normalization (shared by both layers)
  k_init_deg  <<<gN, T, 0, stream>>>(dis);
  k_count_deg <<<gE, T, 0, stream>>>(edst, dis);
  k_rsqrt_deg <<<gN, T, 0, stream>>>(dis);

  // pack weights into WMMA B-fragment layout
  k_pack_w<<<(DD * DD + T - 1) / T, T, 0, stream>>>(W1, Wp);
  k_pack_w<<<(DD * DD + T - 1) / T, T, 0, stream>>>(W2, Wp + (size_t)DD * DD);

  // layer-1 input
  k_concat_bf16<<<gND, T, 0, stream>>>(uemb, bemb, hbf);

  // ---- layer 1 ----
  k_gemm_wmma  <<<gGemm, T, 0, stream>>>(hbf, Wp, hlin);
  k_agg_init   <<<gND, T, 0, stream>>>(hlin, dis, b1, agg);
  k_agg_scatter<<<gScat, T, 0, stream>>>(esrc, edst, hlin, dis, agg);
  k_zero256    <<<1, 256, 0, stream>>>(stats);
  k_bn_stats   <<<512, DD, 0, stream>>>(agg, gsum, gsq);
  k_bn_final   <<<1, DD, 0, stream>>>(gsum, gsq, g1, be1, scale, shift);
  k_bn_relu_bf16<<<gND, T, 0, stream>>>(agg, scale, shift, hbf);

  // ---- layer 2 ----
  k_gemm_wmma  <<<gGemm, T, 0, stream>>>(hbf, Wp + (size_t)DD * DD, hlin);
  k_agg_init   <<<gND, T, 0, stream>>>(hlin, dis, b2, agg);
  k_agg_scatter<<<gScat, T, 0, stream>>>(esrc, edst, hlin, dis, agg);
  k_zero256    <<<1, 256, 0, stream>>>(stats);
  k_bn_stats   <<<512, DD, 0, stream>>>(agg, gsum, gsq);
  k_bn_final   <<<1, DD, 0, stream>>>(gsum, gsq, g2, be2, scale, shift);
  k_bn_relu_f32<<<gND, T, 0, stream>>>(agg, scale, shift, agg);

  // ratings
  k_ratings<<<gRate, T, 0, stream>>>(srcn, dstn, agg, out);
}
